// DepthNet_65360812310652
// MI455X (gfx1250) — matrix operand order repaired and checked by reading, not compiled, not run on hardware
//
#include <hip/hip_runtime.h>
#include <hip/hip_bf16.h>

// Problem constants (fixed by setup_inputs)
#define NV 5
#define NB 2
#define NC 32
#define NH 128
#define NW 160
#define NWP (NW + 2)   // w-padded row for branchless conv halo
#define ND 48

typedef __attribute__((ext_vector_type(16))) _Float16 v16h;
typedef __attribute__((ext_vector_type(8)))  _Float16 v8h;
typedef __attribute__((ext_vector_type(8)))  float    v8f;

union AU { v16h v; v8h h[2]; };

// ---------------------------------------------------------------------------
// K0: combine K*E per (b,v), invert ref proj, store rot(9)+trans(3) per (b,v>=1)
// ---------------------------------------------------------------------------
__device__ inline void combine_proj(const float* p, float* out) {
    // p -> [E(16), K(16)]; out = E with rows0..2 cols0..3 = K[:3,:3] @ E[:3,:4]
    const float* E = p;
    const float* K = p + 16;
    for (int i = 0; i < 16; ++i) out[i] = E[i];
    for (int i = 0; i < 3; ++i)
        for (int j = 0; j < 4; ++j) {
            float s = 0.f;
            for (int k = 0; k < 3; ++k) s += K[i * 4 + k] * E[k * 4 + j];
            out[i * 4 + j] = s;
        }
}

__device__ inline void invert4(const float* m, float* inv) {
    float a[4][8];
    for (int i = 0; i < 4; ++i)
        for (int j = 0; j < 4; ++j) {
            a[i][j]     = m[i * 4 + j];
            a[i][4 + j] = (i == j) ? 1.f : 0.f;
        }
    for (int col = 0; col < 4; ++col) {
        int piv = col;
        float best = fabsf(a[col][col]);
        for (int r = col + 1; r < 4; ++r) {
            float v = fabsf(a[r][col]);
            if (v > best) { best = v; piv = r; }
        }
        if (piv != col)
            for (int j = 0; j < 8; ++j) { float t = a[col][j]; a[col][j] = a[piv][j]; a[piv][j] = t; }
        float p = a[col][col];
        float rp = 1.f / p;
        for (int j = 0; j < 8; ++j) a[col][j] *= rp;
        for (int r = 0; r < 4; ++r) {
            if (r == col) continue;
            float f = a[r][col];
            for (int j = 0; j < 8; ++j) a[r][j] -= f * a[col][j];
        }
    }
    for (int i = 0; i < 4; ++i)
        for (int j = 0; j < 4; ++j) inv[i * 4 + j] = a[i][4 + j];
}

__global__ void k0_prep_proj(const float* __restrict__ proj, float* __restrict__ projw) {
    int t = threadIdx.x;
    if (t >= NB * (NV - 1)) return;
    int b = t / (NV - 1);
    int v = t % (NV - 1) + 1;

    float refc[16], inv[16], srcc[16], P[16];
    combine_proj(proj + (size_t)(b * NV + 0) * 32, refc);
    invert4(refc, inv);
    combine_proj(proj + (size_t)(b * NV + v) * 32, srcc);
    for (int i = 0; i < 4; ++i)
        for (int j = 0; j < 4; ++j) {
            float s = 0.f;
            for (int k = 0; k < 4; ++k) s += srcc[i * 4 + k] * inv[k * 4 + j];
            P[i * 4 + j] = s;
        }
    float* o = projw + (size_t)(b * (NV - 1) + (v - 1)) * 12;
    o[0] = P[0];  o[1] = P[1];  o[2] = P[2];
    o[3] = P[4];  o[4] = P[5];  o[5] = P[6];
    o[6] = P[8];  o[7] = P[9];  o[8] = P[10];
    o[9] = P[3];  o[10] = P[7]; o[11] = P[11];
}

// ---------------------------------------------------------------------------
// K1: transpose features (V,B,C,H,W) -> (V*B, H, W, C)  (channel-last)
// ---------------------------------------------------------------------------
__global__ void k1_transpose(const float* __restrict__ in, float* __restrict__ out) {
    size_t idx = (size_t)blockIdx.x * blockDim.x + threadIdx.x;  // output-linear
    int c = idx % NC;           size_t r = idx / NC;
    int w = r % NW;             r /= NW;
    int h = r % NH;             r /= NH;
    int vb = (int)r;            // 0..V*B-1
    out[idx] = in[((size_t)(vb * NC + c) * NH + h) * NW + w];
}

// ---------------------------------------------------------------------------
// K2: fused warp + mean/var volume.  One (b,d,h) row per block.
//     8 waves/block, wave handles pixels w = waveId + 8*k, lane = channel.
//     Writes f16 channel-last, w-PADDED variance volume (B,D,H,W+2,C);
//     wave 0 zeroes the two border pixels so the conv halo is branchless.
// ---------------------------------------------------------------------------
__global__ void k2_variance(const float* __restrict__ featT,
                            const float* __restrict__ projw,
                            const float* __restrict__ dvals,
                            _Float16* __restrict__ var) {
    int blk = blockIdx.x;                // b*D*H + d*H + h
    int h = blk % NH;
    int t = blk / NH;
    int d = t % ND;
    int b = t / ND;
    int lane = threadIdx.x & 31;         // channel
    int wv   = threadIdx.x >> 5;         // wave id 0..7

    size_t rowBase = (((size_t)b * ND + d) * NH + h) * NWP;   // padded row
    if (wv == 0) {                                             // zero halo cols
        var[(rowBase + 0) * NC + lane]      = (_Float16)0.f;
        var[(rowBase + NW + 1) * NC + lane] = (_Float16)0.f;
    }

    float depth = dvals[b * ND + d];

    float M[(NV - 1) * 12];
    for (int i = 0; i < (NV - 1) * 12; ++i)
        M[i] = projw[(size_t)b * (NV - 1) * 12 + i];

    const float hw = (float)h;
    const float invV = 1.f / (float)NV;

    for (int w = wv; w < NW; w += 8) {
        size_t refIdx = (((size_t)b * NH + h) * NW + w) * NC + lane;  // vb = b (v=0)
        float rf = featT[refIdx];
        float sum = rf;
        float sq  = rf * rf;
        float xw = (float)w;

        for (int v = 1; v < NV; ++v) {
            const float* m = M + (v - 1) * 12;
            float px = (m[0] * xw + m[1] * hw + m[2]) * depth + m[9];
            float py = (m[3] * xw + m[4] * hw + m[5]) * depth + m[10];
            float pz = (m[6] * xw + m[7] * hw + m[8]) * depth + m[11];
            float rz = 1.f / pz;
            float gx = px * rz * (2.f / (float)(NW - 1)) - 1.f;
            float gy = py * rz * (2.f / (float)(NH - 1)) - 1.f;
            float ix = ((gx + 1.f) * (float)NW - 1.f) * 0.5f;
            float iy = ((gy + 1.f) * (float)NH - 1.f) * 0.5f;
            float x0f = floorf(ix), y0f = floorf(iy);
            float wx = ix - x0f, wy = iy - y0f;
            int x0 = (int)x0f, y0 = (int)y0f;

            size_t vbase = (size_t)(v * NB + b) * NH * NW;
            float acc = 0.f;
            int   xs[4] = { x0, x0 + 1, x0, x0 + 1 };
            int   ys[4] = { y0, y0, y0 + 1, y0 + 1 };
            float ws4[4] = { (1.f - wx) * (1.f - wy), wx * (1.f - wy),
                             (1.f - wx) * wy,         wx * wy };
            for (int tap = 0; tap < 4; ++tap) {
                int xi = xs[tap], yi = ys[tap];
                bool valid = (xi >= 0) & (xi <= NW - 1) & (yi >= 0) & (yi <= NH - 1);
                int xc = xi < 0 ? 0 : (xi > NW - 1 ? NW - 1 : xi);
                int yc = yi < 0 ? 0 : (yi > NH - 1 ? NH - 1 : yi);
                float val = featT[(vbase + (size_t)yc * NW + xc) * NC + lane];
                acc += valid ? val * ws4[tap] : 0.f;
            }
            sum += acc;
            sq  += acc * acc;
        }
        float mu = sum * invV;
        float vv = sq * invV - mu * mu;
        var[(rowBase + (w + 1)) * NC + lane] = (_Float16)vv;
    }
}

// ---------------------------------------------------------------------------
// K3: 3D conv (C=32 in, 1 out, 3x3x3, SAME) on padded f16 variance via WMMA.
//     Per wave: M = 16 consecutive w pixels, K = 32 channels, output in N=0.
//     27 chained v_wmma_f32_16x16x32_f16 into one f32 accumulator.
//     Branchless inner loop: 2x ds_load_b128 (B) + 2x global_load_b128 (A).
// ---------------------------------------------------------------------------
#define NTAP 27
__global__ void k3_conv3d(const _Float16* __restrict__ var,
                          const float* __restrict__ w3,
                          const float* __restrict__ bias,
                          float* __restrict__ cost) {
    // wpk[tap][lane][e]: B operand for tap in WMMA layout.
    //   lane L: N = L&15; only N==0 (lanes 0,16) carries weights.
    //   element e of lane L maps to K = (L>>4)*16 + e.
    __shared__ _Float16 wpk[NTAP * 32 * 16];          // 27,648 B of 320 KB WGP LDS
    for (int i = threadIdx.x; i < NTAP * 32 * 16; i += blockDim.x) {
        int e   = i & 15;
        int l   = (i >> 4) & 31;
        int tap = i >> 9;
        float val = 0.f;
        if ((l & 15) == 0) val = w3[(((l >> 4) * 16) + e) * NTAP + tap];
        wpk[i] = (_Float16)val;
    }
    __syncthreads();

    int lane = threadIdx.x & 31;
    // seg is wave-uniform by construction; tell the compiler so the d/h halo
    // checks become scalar branches and EXEC stays all-ones for every WMMA.
    int seg = __builtin_amdgcn_readfirstlane(blockIdx.x * 8 + (threadIdx.x >> 5));
    int sw = seg % (NW / 16);   int t1 = seg / (NW / 16);
    int h  = t1 % NH;           int t2 = t1 / NH;
    int d  = t2 % ND;           int b  = t2 / ND;
    int w0 = sw * 16;

    int m  = lane & 15;             // pixel row within tile
    int kb = (lane >> 4) * 16;      // 0: K 0-15 ; 1: K 16-31

    v8f acc = {};
    for (int kz = 0; kz < 3; ++kz) {
        int dz = d + kz - 1;
        if (dz < 0 || dz >= ND) continue;              // scalar branch
        for (int ky = 0; ky < 3; ++ky) {
            int hy = h + ky - 1;
            if (hy < 0 || hy >= NH) continue;          // scalar branch
            size_t rowBase = (((size_t)b * ND + dz) * NH + hy) * NWP;
            // lane's pixel for kx=0 (padded: +0); kx adds one pixel (NC halves)
            const _Float16* pa = var + (rowBase + (size_t)(w0 + m)) * NC;
            __builtin_prefetch(pa + (size_t)NWP * NC, 0, 1);   // next h row
            for (int kx = 0; kx < 3; ++kx) {
                int tap = kz * 9 + ky * 3 + kx;

                // B: pre-packed, uniform, converged (2x ds_load_b128)
                const _Float16* bp = &wpk[((size_t)tap * 32 + lane) * 16];
                AU bu;
                bu.h[0] = *reinterpret_cast<const v8h*>(bp);
                bu.h[1] = *reinterpret_cast<const v8h*>(bp + 8);

                // A: always in-bounds thanks to zeroed w-halo columns
                const _Float16* p = pa + (size_t)kx * NC;
                AU u;
                u.h[0] = *reinterpret_cast<const v8h*>(p + kb);
                u.h[1] = *reinterpret_cast<const v8h*>(p + kb + 8);

                acc = __builtin_amdgcn_wmma_f32_16x16x32_f16(
                          false, u.v, false, bu.v, (short)0, acc, false, false);
            }
        }
    }

    float bb = bias[0];
    size_t cb = (((size_t)b * ND + d) * NH + h) * NW;
    // D layout: lane 0 -> N=0, M = vgpr; lane 16 -> N=0, M = vgpr+8
    if (lane == 0) {
        for (int r = 0; r < 8; ++r) cost[cb + w0 + r] = acc[r] + bb;
    } else if (lane == 16) {
        for (int r = 0; r < 8; ++r) cost[cb + w0 + 8 + r] = acc[r] + bb;
    }
}

// ---------------------------------------------------------------------------
// K4: softmax over D, depth regression, 4-window confidence, outputs.
// ---------------------------------------------------------------------------
__global__ void k4_finalize(const float* __restrict__ cost,
                            const float* __restrict__ dvals,
                            float* __restrict__ out) {
    int tid = blockIdx.x * blockDim.x + threadIdx.x;   // 0 .. B*H*W-1
    int p = tid % (NH * NW);
    int b = tid / (NH * NW);

    float c[ND], pr[ND];
    float mx = -3.4e38f;
    for (int d = 0; d < ND; ++d) {
        c[d] = cost[((size_t)b * ND + d) * NH * NW + p];
        mx = fmaxf(mx, c[d]);
    }
    float s = 0.f;
    for (int d = 0; d < ND; ++d) { pr[d] = __expf(c[d] - mx); s += pr[d]; }
    float rs = 1.f / s;
    float depth = 0.f, fidx = 0.f;
    for (int d = 0; d < ND; ++d) {
        pr[d] *= rs;
        out[(size_t)(NB * NH * NW) * 2 + ((size_t)b * ND + d) * NH * NW + p] = pr[d];
        depth += pr[d] * dvals[b * ND + d];
        fidx  += pr[d] * (float)d;
    }
    int idx = (int)fidx;
    idx = idx < 0 ? 0 : (idx > ND - 1 ? ND - 1 : idx);
    float conf = 0.f;
    for (int j = idx - 1; j <= idx + 2; ++j)
        if (j >= 0 && j < ND) conf += pr[j];

    out[tid] = depth;
    out[NB * NH * NW + tid] = conf;
    if (tid < NB * ND)
        out[(size_t)(NB * NH * NW) * 2 + (size_t)NB * ND * NH * NW + tid] = dvals[tid];
}

// ---------------------------------------------------------------------------
extern "C" void kernel_launch(void* const* d_in, const int* in_sizes, int n_in,
                              void* d_out, int out_size, void* d_ws, size_t ws_size,
                              hipStream_t stream) {
    (void)in_sizes; (void)n_in; (void)out_size; (void)ws_size;
    const float* features = (const float*)d_in[0];
    const float* proj     = (const float*)d_in[1];
    const float* dvals    = (const float*)d_in[2];
    const float* w3       = (const float*)d_in[3];
    const float* bias     = (const float*)d_in[4];
    float* out = (float*)d_out;

    // Workspace layout (bytes):
    //   projw : 512
    //   featT : V*B*H*W*C * 4        = 26,214,400
    //   var   : B*D*H*(W+2)*C * 2    = 127,401,984   (f16, w-padded; 254,803,968 reserved)
    //   cost  : B*D*H*W * 4          = 7,864,320
    char* ws = (char*)d_ws;
    float*     projw = (float*)ws;
    float*     featT = (float*)(ws + 512);
    _Float16*  var   = (_Float16*)(ws + 512 + 26214400ull);
    float*     cost  = (float*)(ws + 512 + 26214400ull + 254803968ull);

    k0_prep_proj<<<1, 32, 0, stream>>>(proj, projw);
    k1_transpose<<<(NV * NB * NC * NH * NW) / 256, 256, 0, stream>>>(features, featT);
    k2_variance<<<NB * ND * NH, 256, 0, stream>>>(featT, projw, dvals, var);
    k3_conv3d<<<(NB * ND * NH * (NW / 16)) / 8, 256, 0, stream>>>(var, w3, bias, cost);
    k4_finalize<<<(NB * NH * NW) / 256, 256, 0, stream>>>(cost, dvals, out);
}
